// Encoder_1297080123987
// MI455X (gfx1250) — compile-verified, matrix operand order checked by reading, not run on hardware
//
#include <hip/hip_runtime.h>
#include <cstdint>
#include <cstddef>

// ---------------------------------------------------------------------------
// Problem constants (from reference): B=8,W=16,N=64,E=2048
// ---------------------------------------------------------------------------
#define TN 8192      // flattened graph nodes
#define TE 262144    // flattened trace edges (nodes of the line graph)
#define LE 1048576   // line-graph edges
#define DCAT 128
#define DE   64

typedef unsigned int  u32;
typedef unsigned short u16;
typedef __bf16 bf16_t;
typedef bf16_t v16bf __attribute__((ext_vector_type(16)));
typedef float  v8f   __attribute__((ext_vector_type(8)));

// ---------------------------------------------------------------------------
// helpers
// ---------------------------------------------------------------------------
__device__ __forceinline__ u16 f2bf(float x) {
  u32 u = __float_as_uint(x);
  u += 0x7fffu + ((u >> 16) & 1u);      // round-to-nearest-even
  return (u16)(u >> 16);
}

// order-preserving float<->uint mapping for atomicMax on signed floats
__device__ __forceinline__ u32 enc_ord(float f) {
  u32 u = __float_as_uint(f);
  return (u & 0x80000000u) ? ~u : (u | 0x80000000u);
}
__device__ __forceinline__ float dec_ord(u32 m) {
  u32 u = (m & 0x80000000u) ? (m ^ 0x80000000u) : ~m;
  return __uint_as_float(u);
}
#define NEGINF_ORD 0x007FFFFFu   // enc_ord(-inf)

__device__ __forceinline__ float leaky(float v, float s) { return v > 0.f ? v : v * s; }

// Load one 16x32 bf16 WMMA fragment (A layout; B symmetric with Wt=[N,K]):
//   lanes 0-15 : row = base+lane,    K = kk+0..7,  kk+16..23
//   lanes 16-31: row = base+lane-16, K = kk+8..15, kk+24..31
__device__ __forceinline__ v16bf load_frag(const u16* p) {
  union { v16bf v; uint4 u[2]; } f;
  f.u[0] = *(const uint4*)(p);
  f.u[1] = *(const uint4*)(p + 16);
  return f.v;
}

// ---------------------------------------------------------------------------
// WMMA GEMM: C[M,N] = A[M,K](bf16,row-major) @ Wt[N,K]^T(bf16) (+bias)(+leaky)
// Register-blocked: one wave owns a 32x64 output tile (2 A-frags x 4 B-frags
// -> 8 v_wmma per K-step from 12 b128 loads). M%32==0, N%64==0, K%32==0.
// ---------------------------------------------------------------------------
__global__ void k_gemm(const u16* __restrict__ A, const u16* __restrict__ Wt,
                       const float* __restrict__ bias,
                       float* __restrict__ Cf, u16* __restrict__ Cbf,
                       int M, int N, int K, int act, float slope) {
  const int wave = blockIdx.x * (blockDim.x >> 5) + (threadIdx.x >> 5);
  const int lane = threadIdx.x & 31;
  const int wavesN = N >> 6;                  // 64-col blocks
  const int waves  = (M >> 5) * wavesN;       // 32-row blocks
  if (wave >= waves) return;
  const int mb = wave / wavesN, nb = wave % wavesN;

  const int koff = (lane >> 4) << 3;
  const u16* pA0 = A  + (size_t)(mb * 32 + (lane & 15)) * K + koff;
  const u16* pA1 = pA0 + (size_t)16 * K;
  const u16* pB0 = Wt + (size_t)(nb * 64 + (lane & 15)) * K + koff;
  const size_t bstep = (size_t)16 * K;

  v8f acc[2][4] = {};
  for (int kk = 0; kk < K; kk += 32) {
    __builtin_prefetch(pA0 + kk + 32, 0, 0);   // global_prefetch_b8
    __builtin_prefetch(pA1 + kk + 32, 0, 0);
    v16bf a0 = load_frag(pA0 + kk);
    v16bf a1 = load_frag(pA1 + kk);
    v16bf b0 = load_frag(pB0 + kk);
    v16bf b1 = load_frag(pB0 + bstep + kk);
    v16bf b2 = load_frag(pB0 + 2 * bstep + kk);
    v16bf b3 = load_frag(pB0 + 3 * bstep + kk);
    acc[0][0] = __builtin_amdgcn_wmma_f32_16x16x32_bf16(false, a0, false, b0, (short)0, acc[0][0], false, false);
    acc[0][1] = __builtin_amdgcn_wmma_f32_16x16x32_bf16(false, a0, false, b1, (short)0, acc[0][1], false, false);
    acc[0][2] = __builtin_amdgcn_wmma_f32_16x16x32_bf16(false, a0, false, b2, (short)0, acc[0][2], false, false);
    acc[0][3] = __builtin_amdgcn_wmma_f32_16x16x32_bf16(false, a0, false, b3, (short)0, acc[0][3], false, false);
    acc[1][0] = __builtin_amdgcn_wmma_f32_16x16x32_bf16(false, a1, false, b0, (short)0, acc[1][0], false, false);
    acc[1][1] = __builtin_amdgcn_wmma_f32_16x16x32_bf16(false, a1, false, b1, (short)0, acc[1][1], false, false);
    acc[1][2] = __builtin_amdgcn_wmma_f32_16x16x32_bf16(false, a1, false, b2, (short)0, acc[1][2], false, false);
    acc[1][3] = __builtin_amdgcn_wmma_f32_16x16x32_bf16(false, a1, false, b3, (short)0, acc[1][3], false, false);
  }

  // epilogue: VGPR r of tile (i,j) -> row = mb*32+i*16+(lane>=16?8:0)+r,
  //                                   col = nb*64+j*16+(lane&15)
  const int rsub = (lane >> 4) << 3;
  const int csub = lane & 15;
#pragma unroll
  for (int i = 0; i < 2; ++i) {
#pragma unroll
    for (int j = 0; j < 4; ++j) {
      const int colbase = nb * 64 + j * 16 + csub;
      const float bv = bias ? bias[colbase] : 0.f;
#pragma unroll
      for (int r = 0; r < 8; ++r) {
        float v = acc[i][j][r] + bv;
        if (act) v = leaky(v, slope);
        const size_t idx = (size_t)(mb * 32 + i * 16 + rsub + r) * N + colbase;
        if (Cf)  Cf[idx]  = v;
        if (Cbf) Cbf[idx] = f2bf(v);
      }
    }
  }
}

// ---------------------------------------------------------------------------
// elementwise / prep kernels
// ---------------------------------------------------------------------------
__global__ void k_fill_f32(float* p, float v, int n) {
  int i = blockIdx.x * blockDim.x + threadIdx.x;
  if (i < n) p[i] = v;
}
__global__ void k_fill_u32(u32* p, u32 v, int n) {
  int i = blockIdx.x * blockDim.x + threadIdx.x;
  if (i < n) p[i] = v;
}
// node_cat_bf[row,0:64]=x_node, [64:128]=x_log
__global__ void k_concat_bf(const float* __restrict__ xn, const float* __restrict__ xl,
                            u16* __restrict__ out) {
  int i = blockIdx.x * blockDim.x + threadIdx.x;
  if (i >= TN * DCAT) return;
  int row = i >> 7, c = i & 127;
  float v = (c < 64) ? xn[row * 64 + c] : xl[row * 64 + (c - 64)];
  out[i] = f2bf(v);
}
__global__ void k_f32_to_bf(const float* __restrict__ x, u16* __restrict__ out, int n) {
  int i = blockIdx.x * blockDim.x + threadIdx.x;
  if (i < n) out[i] = f2bf(x[i]);
}
// Wt[n,k] = W[k,n] as bf16 (W is [K,N])
__global__ void k_transpose_bf(const float* __restrict__ W, u16* __restrict__ Wt,
                               int K, int N) {
  int i = blockIdx.x * blockDim.x + threadIdx.x;
  if (i >= K * N) return;
  int n = i / K, k = i % K;
  Wt[(size_t)n * K + k] = f2bf(W[(size_t)k * N + n]);
}
// x[row,c] += bias[c]; optional bf16 mirror
__global__ void k_biasadd_bf(float* __restrict__ x, const float* __restrict__ bias,
                             u16* __restrict__ xbf, int rows, int D) {
  int i = blockIdx.x * blockDim.x + threadIdx.x;
  if (i >= rows * D) return;
  float v = x[i] + bias[i % D];
  x[i] = v;
  if (xbf) xbf[i] = f2bf(v);
}

// ---------------------------------------------------------------------------
// GATv2 n2n: scores over 128 cols, 4 heads of 32. One wave per edge.
// ---------------------------------------------------------------------------
__global__ void k_score_n2n(const float* __restrict__ e_n, const float* __restrict__ xl,
                            const float* __restrict__ xr, const int* __restrict__ src,
                            const int* __restrict__ dst, const float* __restrict__ att,
                            float* __restrict__ score) {
  int e = blockIdx.x * (blockDim.x >> 5) + (threadIdx.x >> 5);
  int lane = threadIdx.x & 31;
  if (e >= TE) return;
  size_t s = (size_t)src[e] * DCAT, d = (size_t)dst[e] * DCAT, eb = (size_t)e * DCAT;
  float sc[4];
#pragma unroll
  for (int h = 0; h < 4; ++h) {
    int c = h * 32 + lane;
    float m = e_n[eb + c] + xl[s + c] + xr[d + c];
    sc[h] = leaky(m, 0.2f) * att[h * 32 + lane];
  }
#pragma unroll
  for (int off = 16; off; off >>= 1)
#pragma unroll
    for (int h = 0; h < 4; ++h) sc[h] += __shfl_xor(sc[h], off, 32);
  if (lane == 0)
#pragma unroll
    for (int h = 0; h < 4; ++h) score[(size_t)e * 4 + h] = sc[h];
}

// segment max + exp/segment-sum (shared by both GATs; H=4)
__global__ void k_segmax(const float* __restrict__ score, const int* __restrict__ dst,
                         u32* __restrict__ smax, int nedges) {
  int e = blockIdx.x * blockDim.x + threadIdx.x;
  if (e >= nedges) return;
  int d = dst[e];
#pragma unroll
  for (int h = 0; h < 4; ++h)
    atomicMax(&smax[(size_t)d * 4 + h], enc_ord(score[(size_t)e * 4 + h]));
}
__global__ void k_expsum(const float* __restrict__ score, const int* __restrict__ dst,
                         const u32* __restrict__ smax, float* __restrict__ den,
                         float* __restrict__ ex, int nedges) {
  int e = blockIdx.x * blockDim.x + threadIdx.x;
  if (e >= nedges) return;
  int d = dst[e];
#pragma unroll
  for (int h = 0; h < 4; ++h) {
    float v = __expf(score[(size_t)e * 4 + h] - dec_ord(smax[(size_t)d * 4 + h]));
    ex[(size_t)e * 4 + h] = v;
    atomicAdd(&den[(size_t)d * 4 + h], v);
  }
}

// out[dst] += alpha_h * xl[src]; D=128, head = c/32. One wave per edge.
__global__ void k_agg_n2n(const float* __restrict__ ex, const float* __restrict__ den,
                          const float* __restrict__ xl, const int* __restrict__ src,
                          const int* __restrict__ dst, float* __restrict__ out) {
  int e = blockIdx.x * (blockDim.x >> 5) + (threadIdx.x >> 5);
  int lane = threadIdx.x & 31;
  if (e >= TE) return;
  int s = src[e], d = dst[e];
  float4 exv = *(const float4*)(ex + (size_t)e * 4);
  float4 dnv = *(const float4*)(den + (size_t)d * 4);
  float al[4] = { exv.x / (dnv.x + 1e-16f), exv.y / (dnv.y + 1e-16f),
                  exv.z / (dnv.z + 1e-16f), exv.w / (dnv.w + 1e-16f) };
  size_t sb = (size_t)s * DCAT, db = (size_t)d * DCAT;
#pragma unroll
  for (int h = 0; h < 4; ++h) {
    int c = h * 32 + lane;
    atomicAdd(&out[db + c], al[h] * xl[sb + c]);
  }
}

// ---------------------------------------------------------------------------
// GATv2 e2n: D=64, 4 heads of 16; z (=node2@We) gathered via edge_efea.
// One wave per edge; lane covers c=lane and c=lane+32.
// ---------------------------------------------------------------------------
__global__ void k_score_e2n(const float* __restrict__ xl, const float* __restrict__ xr,
                            const float* __restrict__ z, const int* __restrict__ src,
                            const int* __restrict__ dst, const int* __restrict__ efea,
                            const float* __restrict__ att, float* __restrict__ score) {
  int e = blockIdx.x * (blockDim.x >> 5) + (threadIdx.x >> 5);
  int lane = threadIdx.x & 31;
  if (e >= LE) return;
  size_t s = (size_t)src[e] * DE, d = (size_t)dst[e] * DE, f = (size_t)efea[e] * DE;
  int c0 = lane, c1 = lane + 32;
  float m0 = leaky(xl[s + c0] + xr[d + c0] + z[f + c0], 0.2f);
  float m1 = leaky(xl[s + c1] + xr[d + c1] + z[f + c1], 0.2f);
  int hg = lane >> 4, cc = lane & 15;
  float v0 = m0 * att[hg * 16 + cc];         // heads 0/1
  float v1 = m1 * att[(2 + hg) * 16 + cc];   // heads 2/3
#pragma unroll
  for (int off = 8; off; off >>= 1) {        // reduce within 16-lane groups
    v0 += __shfl_xor(v0, off, 32);
    v1 += __shfl_xor(v1, off, 32);
  }
  if (cc == 0) {
    score[(size_t)e * 4 + hg]     = v0;
    score[(size_t)e * 4 + 2 + hg] = v1;
  }
}

__global__ void k_agg_e2n(const float* __restrict__ ex, const float* __restrict__ den,
                          const float* __restrict__ xl, const int* __restrict__ src,
                          const int* __restrict__ dst, float* __restrict__ out) {
  int e = blockIdx.x * (blockDim.x >> 5) + (threadIdx.x >> 5);
  int lane = threadIdx.x & 31;
  if (e >= LE) return;
  int s = src[e], d = dst[e];
  float4 exv = *(const float4*)(ex + (size_t)e * 4);
  float4 dnv = *(const float4*)(den + (size_t)d * 4);
  float al[4] = { exv.x / (dnv.x + 1e-16f), exv.y / (dnv.y + 1e-16f),
                  exv.z / (dnv.z + 1e-16f), exv.w / (dnv.w + 1e-16f) };
  size_t sb = (size_t)s * DE, db = (size_t)d * DE;
  int c0 = lane, c1 = lane + 32, hg = lane >> 4;
  atomicAdd(&out[db + c0], al[hg]     * xl[sb + c0]);
  atomicAdd(&out[db + c1], al[2 + hg] * xl[sb + c1]);
}

// ---------------------------------------------------------------------------
// AddNorm over D=64: one wave per row, 2 cols per lane, shfl reductions.
// y has stride/offset so node2[:, :64] and node2[:, 64:] can alias it.
// ---------------------------------------------------------------------------
__global__ void k_add_ln(const float* __restrict__ y, int ystride, int yoff,
                         const float* __restrict__ res, const float* __restrict__ g,
                         const float* __restrict__ b, float* __restrict__ outf,
                         u16* __restrict__ outbf, int rows) {
  int row = blockIdx.x * (blockDim.x >> 5) + (threadIdx.x >> 5);
  int lane = threadIdx.x & 31;
  if (row >= rows) return;
  const float* yr = y + (size_t)row * ystride + yoff;
  const float* rr = res + (size_t)row * 64;
  float v0 = yr[lane] + rr[lane];
  float v1 = yr[lane + 32] + rr[lane + 32];
  float s = v0 + v1;
#pragma unroll
  for (int o = 16; o; o >>= 1) s += __shfl_xor(s, o, 32);
  float mu = s * (1.f / 64.f);
  float d0 = v0 - mu, d1 = v1 - mu;
  float q = d0 * d0 + d1 * d1;
#pragma unroll
  for (int o = 16; o; o >>= 1) q += __shfl_xor(q, o, 32);
  float rs = rsqrtf(q * (1.f / 64.f) + 1e-5f);
  float o0 = d0 * rs * g[lane] + b[lane];
  float o1 = d1 * rs * g[lane + 32] + b[lane + 32];
  size_t ob = (size_t)row * 64;
  outf[ob + lane] = o0;
  outf[ob + lane + 32] = o1;
  if (outbf) { outbf[ob + lane] = f2bf(o0); outbf[ob + lane + 32] = f2bf(o1); }
}

// ---------------------------------------------------------------------------
// host driver
// ---------------------------------------------------------------------------
static inline int ceil_div(long long a, int b) { return (int)((a + b - 1) / b); }

static void gemm(hipStream_t st, const u16* A, const u16* Wt, const float* bias,
                 float* Cf, u16* Cbf, int M, int N, int K, int act, float slope) {
  int tiles = (M / 32) * (N / 64);   // one wave per 32x64 tile
  k_gemm<<<ceil_div(tiles, 8), 256, 0, st>>>(A, Wt, bias, Cf, Cbf, M, N, K, act, slope);
}

extern "C" void kernel_launch(void* const* d_in, const int* in_sizes, int n_in,
                              void* d_out, int out_size, void* d_ws, size_t ws_size,
                              hipStream_t stream) {
  (void)in_sizes; (void)n_in; (void)out_size; (void)ws_size;
  // ---- inputs (setup_inputs dict order, params flattened in dict order) ----
  const float* x_node = (const float*)d_in[0];   // [TN,64]
  const float* x_trace = (const float*)d_in[1];  // [TE,64]
  const float* x_log  = (const float*)d_in[2];   // [TN,64]
  const int* node_adj = (const int*)d_in[3];     // [2,TE]
  const int* edge_adj = (const int*)d_in[4];     // [2,LE]
  const int* edge_efea = (const int*)d_in[5];    // [LE]
  const float* Wl_n = (const float*)d_in[6],  *bl_n = (const float*)d_in[7];
  const float* Wr_n = (const float*)d_in[8],  *br_n = (const float*)d_in[9];
  const float* We_n = (const float*)d_in[10], *att_n = (const float*)d_in[11];
  const float* bias_n = (const float*)d_in[12];
  const float* Wl_e = (const float*)d_in[13], *bl_e = (const float*)d_in[14];
  const float* Wr_e = (const float*)d_in[15], *br_e = (const float*)d_in[16];
  const float* We_e = (const float*)d_in[17], *att_e = (const float*)d_in[18];
  const float* bias_e = (const float*)d_in[19];
  const float* g1n = (const float*)d_in[20], *b1n = (const float*)d_in[21];
  const float* g1t = (const float*)d_in[22], *b1t = (const float*)d_in[23];
  const float* g1l = (const float*)d_in[24], *b1l = (const float*)d_in[25];
  const float* W1n = (const float*)d_in[26], *b1fn = (const float*)d_in[27];
  const float* W2n = (const float*)d_in[28], *b2fn = (const float*)d_in[29];
  const float* W1t = (const float*)d_in[30], *b1ft = (const float*)d_in[31];
  const float* W2t = (const float*)d_in[32], *b2ft = (const float*)d_in[33];
  const float* W1l = (const float*)d_in[34], *b1fl = (const float*)d_in[35];
  const float* W2l = (const float*)d_in[36], *b2fl = (const float*)d_in[37];
  const float* g2n = (const float*)d_in[38], *b2n = (const float*)d_in[39];
  const float* g2t = (const float*)d_in[40], *b2t = (const float*)d_in[41];
  const float* g2l = (const float*)d_in[42], *b2l = (const float*)d_in[43];

  const int* src_n = node_adj,       *dst_n = node_adj + TE;
  const int* src_e = edge_adj,       *dst_e = edge_adj + LE;

  // ---- workspace bump allocator ----
  char* wsp = (char*)d_ws;
  size_t off = 0;
  auto carve = [&](size_t bytes) -> void* {
    void* p = wsp + off;
    off = (off + bytes + 255) & ~(size_t)255;
    return p;
  };
  u16* node_cat_bf = (u16*)carve((size_t)TN * 128 * 2);
  u16* trace_bf    = (u16*)carve((size_t)TE * 64 * 2);
  u16* WlT_n = (u16*)carve(128 * 128 * 2);
  u16* WrT_n = (u16*)carve(128 * 128 * 2);
  u16* WeT_n = (u16*)carve(128 * 64 * 2);   // [N=128,K=64]
  u16* WlT_e = (u16*)carve(64 * 64 * 2);
  u16* WrT_e = (u16*)carve(64 * 64 * 2);
  u16* WeT_e = (u16*)carve(64 * 128 * 2);   // [N=64,K=128]
  u16* W1T_n = (u16*)carve(256 * 64 * 2);
  u16* W2T_n = (u16*)carve(64 * 256 * 2);
  u16* W1T_t = (u16*)carve(256 * 64 * 2);
  u16* W2T_t = (u16*)carve(64 * 256 * 2);
  u16* W1T_l = (u16*)carve(256 * 64 * 2);
  u16* W2T_l = (u16*)carve(64 * 256 * 2);
  float* xl_n  = (float*)carve((size_t)TN * 128 * 4);
  float* xr_n  = (float*)carve((size_t)TN * 128 * 4);
  float* e_n   = (float*)carve((size_t)TE * 128 * 4);   // 128MB, recycled below
  float* score_n = (float*)carve((size_t)TE * 4 * 4);
  float* ex_n  = (float*)carve((size_t)TE * 4 * 4);
  u32*   smax_n = (u32*)carve((size_t)TN * 4 * 4);
  float* den_n = (float*)carve((size_t)TN * 4 * 4);
  float* node2 = (float*)carve((size_t)TN * 128 * 4);
  u16*  node2_bf = (u16*)carve((size_t)TN * 128 * 2);
  float* z = (float*)carve((size_t)TN * 64 * 4);
  float* score_t = (float*)carve((size_t)LE * 4 * 4);
  float* ex_t  = (float*)carve((size_t)LE * 4 * 4);
  u32*   smax_t = (u32*)carve((size_t)TE * 4 * 4);
  float* den_t = (float*)carve((size_t)TE * 4 * 4);
  float* trace2 = (float*)carve((size_t)TE * 64 * 4);
  float* xn1 = (float*)carve((size_t)TN * 64 * 4);
  u16*  xn1_bf = (u16*)carve((size_t)TN * 64 * 2);
  float* xl1 = (float*)carve((size_t)TN * 64 * 4);
  u16*  xl1_bf = (u16*)carve((size_t)TN * 64 * 2);
  float* xt1 = (float*)carve((size_t)TE * 64 * 4);
  u16*  xt1_bf = (u16*)carve((size_t)TE * 64 * 2);
  u16*  hid_n = (u16*)carve((size_t)TN * 256 * 2);
  u16*  hid_l = (u16*)carve((size_t)TN * 256 * 2);
  u16*  hid_t = (u16*)carve((size_t)TE * 256 * 2);
  float* ffo_n = (float*)carve((size_t)TN * 64 * 4);
  float* ffo_l = (float*)carve((size_t)TN * 64 * 4);
  float* ffo_t = (float*)carve((size_t)TE * 64 * 4);
  // recycle e_n (dead after n2n scores) for e2n xl/xr
  float* xl_t = e_n;
  float* xr_t = e_n + (size_t)TE * 64;

  float* out_xn = (float*)d_out;                       // [TN,64]
  float* out_xt = out_xn + (size_t)TN * 64;            // [TE,64]
  float* out_xl = out_xt + (size_t)TE * 64;            // [TN,64]

  const int T = 256;
  // ---- prep: conversions / transposes / clears ----
  k_concat_bf<<<ceil_div((size_t)TN * 128, T), T, 0, stream>>>(x_node, x_log, node_cat_bf);
  k_f32_to_bf<<<ceil_div((size_t)TE * 64, T), T, 0, stream>>>(x_trace, trace_bf, TE * 64);
  k_transpose_bf<<<ceil_div(128 * 128, T), T, 0, stream>>>(Wl_n, WlT_n, 128, 128);
  k_transpose_bf<<<ceil_div(128 * 128, T), T, 0, stream>>>(Wr_n, WrT_n, 128, 128);
  k_transpose_bf<<<ceil_div(64 * 128, T),  T, 0, stream>>>(We_n, WeT_n, 64, 128);
  k_transpose_bf<<<ceil_div(64 * 64, T),   T, 0, stream>>>(Wl_e, WlT_e, 64, 64);
  k_transpose_bf<<<ceil_div(64 * 64, T),   T, 0, stream>>>(Wr_e, WrT_e, 64, 64);
  k_transpose_bf<<<ceil_div(128 * 64, T),  T, 0, stream>>>(We_e, WeT_e, 128, 64);
  k_transpose_bf<<<ceil_div(64 * 256, T),  T, 0, stream>>>(W1n, W1T_n, 64, 256);
  k_transpose_bf<<<ceil_div(256 * 64, T),  T, 0, stream>>>(W2n, W2T_n, 256, 64);
  k_transpose_bf<<<ceil_div(64 * 256, T),  T, 0, stream>>>(W1t, W1T_t, 64, 256);
  k_transpose_bf<<<ceil_div(256 * 64, T),  T, 0, stream>>>(W2t, W2T_t, 256, 64);
  k_transpose_bf<<<ceil_div(64 * 256, T),  T, 0, stream>>>(W1l, W1T_l, 64, 256);
  k_transpose_bf<<<ceil_div(256 * 64, T),  T, 0, stream>>>(W2l, W2T_l, 256, 64);
  k_fill_f32<<<ceil_div((size_t)TN * 128, T), T, 0, stream>>>(node2, 0.f, TN * 128);
  k_fill_f32<<<ceil_div((size_t)TE * 64, T),  T, 0, stream>>>(trace2, 0.f, TE * 64);
  k_fill_f32<<<ceil_div(TN * 4, T), T, 0, stream>>>(den_n, 0.f, TN * 4);
  k_fill_f32<<<ceil_div(TE * 4, T), T, 0, stream>>>(den_t, 0.f, TE * 4);
  k_fill_u32<<<ceil_div(TN * 4, T), T, 0, stream>>>(smax_n, NEGINF_ORD, TN * 4);
  k_fill_u32<<<ceil_div(TE * 4, T), T, 0, stream>>>(smax_t, NEGINF_ORD, TE * 4);

  // ---- GATv2 node2node ----
  gemm(stream, node_cat_bf, WlT_n, bl_n, xl_n, nullptr, TN, 128, 128, 0, 0.f);
  gemm(stream, node_cat_bf, WrT_n, br_n, xr_n, nullptr, TN, 128, 128, 0, 0.f);
  gemm(stream, trace_bf,    WeT_n, nullptr, e_n, nullptr, TE, 128, 64, 0, 0.f);
  k_score_n2n<<<ceil_div(TE, 8), 256, 0, stream>>>(e_n, xl_n, xr_n, src_n, dst_n, att_n, score_n);
  k_segmax<<<ceil_div(TE, T), T, 0, stream>>>(score_n, dst_n, smax_n, TE);
  k_expsum<<<ceil_div(TE, T), T, 0, stream>>>(score_n, dst_n, smax_n, den_n, ex_n, TE);
  k_agg_n2n<<<ceil_div(TE, 8), 256, 0, stream>>>(ex_n, den_n, xl_n, src_n, dst_n, node2);
  k_biasadd_bf<<<ceil_div((size_t)TN * 128, T), T, 0, stream>>>(node2, bias_n, node2_bf, TN, 128);

  // ---- GATv2 edge2node (line graph); We hoisted before the efea gather ----
  gemm(stream, trace_bf, WlT_e, bl_e, xl_t, nullptr, TE, 64, 64, 0, 0.f);
  gemm(stream, trace_bf, WrT_e, br_e, xr_t, nullptr, TE, 64, 64, 0, 0.f);
  gemm(stream, node2_bf, WeT_e, nullptr, z, nullptr, TN, 64, 128, 0, 0.f);
  k_score_e2n<<<ceil_div(LE, 8), 256, 0, stream>>>(xl_t, xr_t, z, src_e, dst_e, edge_efea, att_e, score_t);
  k_segmax<<<ceil_div(LE, T), T, 0, stream>>>(score_t, dst_e, smax_t, LE);
  k_expsum<<<ceil_div(LE, T), T, 0, stream>>>(score_t, dst_e, smax_t, den_t, ex_t, LE);
  k_agg_e2n<<<ceil_div(LE, 8), 256, 0, stream>>>(ex_t, den_t, xl_t, src_e, dst_e, trace2);
  k_biasadd_bf<<<ceil_div((size_t)TE * 64, T), T, 0, stream>>>(trace2, bias_e, nullptr, TE, 64);

  // ---- AddNorm 1 ----
  k_add_ln<<<ceil_div(TN, 8), 256, 0, stream>>>(node2, 128, 0,  x_node,  g1n, b1n, xn1, xn1_bf, TN);
  k_add_ln<<<ceil_div(TN, 8), 256, 0, stream>>>(node2, 128, 64, x_log,   g1l, b1l, xl1, xl1_bf, TN);
  k_add_ln<<<ceil_div(TE, 8), 256, 0, stream>>>(trace2, 64, 0,  x_trace, g1t, b1t, xt1, xt1_bf, TE);

  // ---- FFN + AddNorm 2 (outputs straight into d_out) ----
  gemm(stream, xn1_bf, W1T_n, b1fn, nullptr, hid_n, TN, 256, 64, 1, 0.01f);
  gemm(stream, hid_n,  W2T_n, b2fn, ffo_n, nullptr, TN, 64, 256, 0, 0.f);
  k_add_ln<<<ceil_div(TN, 8), 256, 0, stream>>>(ffo_n, 64, 0, xn1, g2n, b2n, out_xn, nullptr, TN);

  gemm(stream, xt1_bf, W1T_t, b1ft, nullptr, hid_t, TE, 256, 64, 1, 0.01f);
  gemm(stream, hid_t,  W2T_t, b2ft, ffo_t, nullptr, TE, 64, 256, 0, 0.f);
  k_add_ln<<<ceil_div(TE, 8), 256, 0, stream>>>(ffo_t, 64, 0, xt1, g2t, b2t, out_xt, nullptr, TE);

  gemm(stream, xl1_bf, W1T_l, b1fl, nullptr, hid_l, TN, 256, 64, 1, 0.01f);
  gemm(stream, hid_l,  W2T_l, b2fl, ffo_l, nullptr, TN, 64, 256, 0, 0.f);
  k_add_ln<<<ceil_div(TN, 8), 256, 0, stream>>>(ffo_l, 64, 0, xl1, g2l, b2l, out_xl, nullptr, TN);
}
